// GeologyAwareAttention_38113539785274
// MI455X (gfx1250) — compile-verified, matrix operand order checked by reading, not run on hardware
//
#include <hip/hip_runtime.h>
#include <hip/hip_bf16.h>
#include <math.h>

#define B_  4
#define S_  2048
#define D_  1024
#define H_  16
#define HD_ 64
#define G_  32

typedef _Float16 v16h __attribute__((ext_vector_type(16)));
typedef _Float16 v8h  __attribute__((ext_vector_type(8)));
typedef float    v8f  __attribute__((ext_vector_type(8)));

__device__ __forceinline__ v8f wmma_f16f32(v16h a, v16h b, v8f c) {
  // D = A(16x32 f16) * B(32x16 f16) + C(16x16 f32)
  return __builtin_amdgcn_wmma_f32_16x16x32_f16(false, a, false, b, (short)0, c,
                                                false, false);
}

// Load a 16x32 f16 fragment (A-layout per CDNA5 ISA 7.12.2) from a row-major
// matrix. lane -> row = rowbase + (lane&15); halves 0..7 cover k=k0+hf*8..+7,
// halves 8..15 cover k=k0+16+hf*8..+7  (hf = lane>>4). Two global_load_b128.
__device__ __forceinline__ v16h load_frag(const _Float16* __restrict__ base,
                                          int row, int ld, int k0, int hf) {
  const _Float16* p = base + (size_t)row * ld + k0 + hf * 8;
  v8h lo = *(const v8h*)(p);
  v8h hi = *(const v8h*)(p + 16);
  v16h a;
#pragma unroll
  for (int i = 0; i < 8; ++i) { a[i] = lo[i]; a[i + 8] = hi[i]; }
  return a;
}

// ---------------------------------------------------------------- converts
__global__ __launch_bounds__(256) void cvt_f16(const float* __restrict__ src,
                                               _Float16* __restrict__ dst, int n) {
  int i = blockIdx.x * 256 + threadIdx.x;
  if (i < n) dst[i] = (_Float16)src[i];
}

// W (K x N, row-major f32) -> Wt (N x K, row-major f16)
__global__ __launch_bounds__(256) void transpose_cvt(const float* __restrict__ W,
                                                     _Float16* __restrict__ Wt,
                                                     int K, int N) {
  int i = blockIdx.x * 256 + threadIdx.x;
  if (i < K * N) {
    int k = i / N, n = i - k * N;
    Wt[(size_t)n * K + k] = (_Float16)W[i];
  }
}

// ---------------------------------------------------------------- geo path
__global__ __launch_bounds__(256) void geo1_ln_relu(
    const float* __restrict__ geo, const float* __restrict__ W1,
    const float* __restrict__ b1, const float* __restrict__ lng,
    const float* __restrict__ lnb, float* __restrict__ hbuf) {
  int b = blockIdx.x, tid = threadIdx.x;
  __shared__ float sg[G_];
  __shared__ float red[256];
  if (tid < G_) sg[tid] = geo[b * G_ + tid];
  __syncthreads();
  float v[4];
#pragma unroll
  for (int j = 0; j < 4; ++j) {
    int d = tid + j * 256;
    float acc = b1[d];
    for (int g = 0; g < G_; ++g) acc += sg[g] * W1[g * D_ + d];
    v[j] = acc;
  }
  float s = v[0] + v[1] + v[2] + v[3];
  red[tid] = s; __syncthreads();
  for (int o = 128; o > 0; o >>= 1) { if (tid < o) red[tid] += red[tid + o]; __syncthreads(); }
  float mu = red[0] * (1.0f / D_);
  __syncthreads();
  float s2 = 0.f;
#pragma unroll
  for (int j = 0; j < 4; ++j) { float d0 = v[j] - mu; s2 += d0 * d0; }
  red[tid] = s2; __syncthreads();
  for (int o = 128; o > 0; o >>= 1) { if (tid < o) red[tid] += red[tid + o]; __syncthreads(); }
  float inv = rsqrtf(red[0] * (1.0f / D_) + 1e-5f);
#pragma unroll
  for (int j = 0; j < 4; ++j) {
    int d = tid + j * 256;
    float t = (v[j] - mu) * inv * lng[d] + lnb[d];
    hbuf[b * D_ + d] = fmaxf(t, 0.f);
  }
}

__global__ __launch_bounds__(256) void geo2_linear(
    const float* __restrict__ hbuf, const float* __restrict__ W2,
    const float* __restrict__ b2, float* __restrict__ enc) {
  int i = blockIdx.x * 256 + threadIdx.x;  // i < B_*D_
  int b = i >> 10, d = i & (D_ - 1);
  float acc = b2[d];
  for (int k = 0; k < D_; ++k) acc += hbuf[b * D_ + k] * W2[k * D_ + d];
  enc[i] = acc;
}

__global__ __launch_bounds__(64) void geo3_gate(
    const float* __restrict__ enc, const float* __restrict__ gmW,
    const float* __restrict__ gmb, float* __restrict__ gmod) {
  int i = threadIdx.x;  // i < B_*H_
  int b = i >> 4, h = i & (H_ - 1);
  float acc = gmb[h];
  for (int k = 0; k < D_; ++k) acc += enc[b * D_ + k] * gmW[k * H_ + h];
  gmod[i] = 1.0f / (1.0f + __expf(-acc));
}

// ---------------------------------------------------------------- WMMA GEMM
// C(MxN) = A(MxK f16 rowmajor) * BT(NxK f16 rowmajor)^T + bias
// MODE 1: f16 split-heads (B,H,S,HD); MODE 2: f16 V-transposed (B,H,HD,S);
// MODE 3: f32 row-major to d_out.
template <int MODE>
__global__ __launch_bounds__(256) void gemm_wmma(
    const _Float16* __restrict__ A, const _Float16* __restrict__ BT,
    const float* __restrict__ bias, void* __restrict__ Cout,
    int M, int N, int K) {
  int tid = threadIdx.x;
  int wv = tid >> 5, lane = tid & 31;
  int hf = lane >> 4, lr = lane & 15;
  int m0 = blockIdx.y * 128 + wv * 16;
  int n0 = blockIdx.x * 64;
  v8f acc[4] = {};
  for (int k0 = 0; k0 < K; k0 += 32) {
    v16h a = load_frag(A, m0 + lr, K, k0, hf);
#pragma unroll
    for (int t = 0; t < 4; ++t) {
      v16h b = load_frag(BT, n0 + t * 16 + lr, K, k0, hf);
      acc[t] = wmma_f16f32(a, b, acc[t]);
    }
  }
#pragma unroll
  for (int t = 0; t < 4; ++t) {
#pragma unroll
    for (int r = 0; r < 8; ++r) {
      int row = m0 + r + 8 * hf;          // global row in [0, B*S)
      int col = n0 + t * 16 + lr;         // global col in [0, D)
      float v = acc[t][r] + bias[col];
      if (MODE == 3) {
        ((float*)Cout)[(size_t)row * N + col] = v;
      } else {
        int bb = row >> 11, s = row & (S_ - 1);
        int h = col >> 6, hd = col & (HD_ - 1);
        size_t bh = (size_t)bb * H_ + h;
        if (MODE == 1)
          ((_Float16*)Cout)[(bh * S_ + s) * HD_ + hd] = (_Float16)v;
        else
          ((_Float16*)Cout)[(bh * HD_ + hd) * S_ + s] = (_Float16)v;
      }
    }
  }
}

// ---------------------------------------------------------------- attention
// One block = (b, h, 16 query rows). 8 waves. Scores strip 16x2048 lives in
// LDS (128 KB); attn written to HBM exactly once; ctx computed from LDS.
__global__ __launch_bounds__(256) void attn_fused(
    const _Float16* __restrict__ Qh, const _Float16* __restrict__ Kh,
    const _Float16* __restrict__ Vt, const float* __restrict__ gmod,
    float* __restrict__ attn_out, _Float16* __restrict__ ctx_out) {
  extern __shared__ float smem[];
  float* s_sc     = smem;               // 16*2048 f32 scores / reused for parts
  float* s_red    = smem + 16 * S_;     // 256
  float* s_rowmax = s_red + 256;        // 16
  float* s_rowsum = s_rowmax + 16;      // 16
  float* s_rowinv = s_rowsum + 16;      // 16

  int tid = threadIdx.x;
  int wv = tid >> 5, lane = tid & 31;
  int hf = lane >> 4, lr = lane & 15;
  int strip = blockIdx.x;
  int m0 = (strip & 127) * 16;
  int bh = strip >> 7;

  const _Float16* Qb = Qh + (size_t)bh * S_ * HD_;
  const _Float16* Kb = Kh + (size_t)bh * S_ * HD_;
  const _Float16* Vb = Vt + (size_t)bh * HD_ * S_;
  float cs = 0.125f * (1.0f + gmod[bh]);  // 1/sqrt(64) * (1+geo_mod)

  // ---- scores: Q(16x64) * K^T tiles, each wave owns 16 of 128 n-tiles
  v16h aq0 = load_frag(Qb, m0 + lr, HD_, 0, hf);
  v16h aq1 = load_frag(Qb, m0 + lr, HD_, 32, hf);
  for (int nt = wv * 16; nt < wv * 16 + 16; ++nt) {
    int n0 = nt * 16;
    v16h bk0 = load_frag(Kb, n0 + lr, HD_, 0, hf);
    v16h bk1 = load_frag(Kb, n0 + lr, HD_, 32, hf);
    v8f c = {};
    c = wmma_f16f32(aq0, bk0, c);
    c = wmma_f16f32(aq1, bk1, c);
#pragma unroll
    for (int r = 0; r < 8; ++r)
      s_sc[(r + 8 * hf) * S_ + n0 + lr] = c[r] * cs;
  }
  __syncthreads();

  // ---- softmax over each of the 16 rows (16 threads per row)
  int row = tid >> 4, chunk = tid & 15;
  float* rowp = s_sc + row * S_ + chunk * 128;
  float mx = -1e30f;
  for (int i = 0; i < 128; ++i) mx = fmaxf(mx, rowp[i]);
  s_red[row * 16 + chunk] = mx;
  __syncthreads();
  if (tid < 16) {
    float m = -1e30f;
    for (int j = 0; j < 16; ++j) m = fmaxf(m, s_red[tid * 16 + j]);
    s_rowmax[tid] = m;
  }
  __syncthreads();
  float rm = s_rowmax[row];
  float sum = 0.f;
  for (int i = 0; i < 128; ++i) {
    float e = __expf(rowp[i] - rm);
    rowp[i] = e;
    sum += e;
  }
  s_red[row * 16 + chunk] = sum;
  __syncthreads();
  if (tid < 16) {
    float s = 0.f;
    for (int j = 0; j < 16; ++j) s += s_red[tid * 16 + j];
    s_rowsum[tid] = s;
    s_rowinv[tid] = 1.0f / s;
  }
  __syncthreads();

  // ---- single coalesced normalized attn write (dominant HBM stream)
  {
    size_t abase = ((size_t)bh * S_ + m0) * S_;
    for (int idx = tid; idx < 16 * S_; idx += 256)
      attn_out[abase + idx] = s_sc[idx] * s_rowinv[idx >> 11];
  }

  // ---- ctx = (e @ V) * rowinv, K split across waves (8 k-steps each)
  v8f cacc[4] = {};
  for (int ks = wv * 8; ks < wv * 8 + 8; ++ks) {
    int k0 = ks * 32;
    const float* ap = s_sc + lr * S_ + k0 + hf * 8;
    v16h a;
#pragma unroll
    for (int i = 0; i < 8; ++i) { a[i] = (_Float16)ap[i]; a[i + 8] = (_Float16)ap[i + 16]; }
#pragma unroll
    for (int t = 0; t < 4; ++t) {
      v16h bv = load_frag(Vb, t * 16 + lr, S_, k0, hf);
      cacc[t] = wmma_f16f32(a, bv, cacc[t]);
    }
  }
  __syncthreads();             // all reads of s_sc complete before aliasing
  float* s_part = s_sc;        // reuse as [8 waves][16*64]
#pragma unroll
  for (int t = 0; t < 4; ++t)
#pragma unroll
    for (int r = 0; r < 8; ++r)
      s_part[wv * 1024 + (r + 8 * hf) * 64 + t * 16 + lr] = cacc[t][r];
  __syncthreads();
  int b = bh >> 4, h = bh & (H_ - 1);
  for (int idx = tid; idx < 1024; idx += 256) {
    float s = 0.f;
#pragma unroll
    for (int w = 0; w < 8; ++w) s += s_part[w * 1024 + idx];
    int rr = idx >> 6, cc = idx & 63;
    s *= s_rowinv[rr];
    ctx_out[((size_t)b * S_ + m0 + rr) * D_ + h * HD_ + cc] = (_Float16)s;
  }
}

// ---------------------------------------------------------------- launch
extern "C" void kernel_launch(void* const* d_in, const int* in_sizes, int n_in,
                              void* d_out, int out_size, void* d_ws, size_t ws_size,
                              hipStream_t stream) {
  (void)in_sizes; (void)n_in; (void)out_size; (void)ws_size;
  const float* x    = (const float*)d_in[0];
  const float* geo  = (const float*)d_in[1];
  const float* Wq   = (const float*)d_in[2];  const float* bq  = (const float*)d_in[3];
  const float* Wk   = (const float*)d_in[4];  const float* bk  = (const float*)d_in[5];
  const float* Wv   = (const float*)d_in[6];  const float* bv  = (const float*)d_in[7];
  const float* Wo   = (const float*)d_in[8];  const float* bo  = (const float*)d_in[9];
  const float* geW1 = (const float*)d_in[10]; const float* geb1 = (const float*)d_in[11];
  const float* lng  = (const float*)d_in[12]; const float* lnb  = (const float*)d_in[13];
  const float* geW2 = (const float*)d_in[14]; const float* geb2 = (const float*)d_in[15];
  const float* gmW  = (const float*)d_in[16]; const float* gmb  = (const float*)d_in[17];

  char* ws = (char*)d_ws;
  size_t off = 0;
  auto alloc = [&](size_t bytes) -> void* {
    void* p = ws + off;
    off = (off + bytes + 255) & ~(size_t)255;
    return p;
  };
  const size_t MD = (size_t)B_ * S_ * D_;           // 8,388,608
  _Float16* xh   = (_Float16*)alloc(MD * 2);
  _Float16* WqT  = (_Float16*)alloc((size_t)D_ * D_ * 2);
  _Float16* WkT  = (_Float16*)alloc((size_t)D_ * D_ * 2);
  _Float16* WvT  = (_Float16*)alloc((size_t)D_ * D_ * 2);
  _Float16* WoT  = (_Float16*)alloc((size_t)D_ * D_ * 2);
  _Float16* Qh   = (_Float16*)alloc(MD * 2);
  _Float16* Kh   = (_Float16*)alloc(MD * 2);
  _Float16* Vth  = (_Float16*)alloc(MD * 2);
  _Float16* ctxh = (_Float16*)alloc(MD * 2);
  float* hbuf = (float*)alloc((size_t)B_ * D_ * 4);
  float* enc  = (float*)alloc((size_t)B_ * D_ * 4);
  float* gmod = (float*)alloc((size_t)B_ * H_ * 4);

  // converts
  cvt_f16<<<(int)(MD / 256), 256, 0, stream>>>(x, xh, (int)MD);
  int wn = D_ * D_;
  transpose_cvt<<<wn / 256, 256, 0, stream>>>(Wq, WqT, D_, D_);
  transpose_cvt<<<wn / 256, 256, 0, stream>>>(Wk, WkT, D_, D_);
  transpose_cvt<<<wn / 256, 256, 0, stream>>>(Wv, WvT, D_, D_);
  transpose_cvt<<<wn / 256, 256, 0, stream>>>(Wo, WoT, D_, D_);

  // geo path
  geo1_ln_relu<<<B_, 256, 0, stream>>>(geo, geW1, geb1, lng, lnb, hbuf);
  geo2_linear<<<(B_ * D_) / 256, 256, 0, stream>>>(hbuf, geW2, geb2, enc);
  geo3_gate<<<1, 64, 0, stream>>>(enc, gmW, gmb, gmod);

  // projections
  dim3 ggrid(D_ / 64, (B_ * S_) / 128);
  gemm_wmma<1><<<ggrid, 256, 0, stream>>>(xh, WqT, bq, Qh, B_ * S_, D_, D_);
  gemm_wmma<1><<<ggrid, 256, 0, stream>>>(xh, WkT, bk, Kh, B_ * S_, D_, D_);
  gemm_wmma<2><<<ggrid, 256, 0, stream>>>(xh, WvT, bv, Vth, B_ * S_, D_, D_);

  // fused attention (attn goes to d_out after the main output block)
  float* attn_out = (float*)d_out + MD;
  int smem_bytes = (16 * S_ + 256 + 16 + 16 + 16) * 4;  // 132,288 B
  (void)hipFuncSetAttribute((const void*)attn_fused,
                            hipFuncAttributeMaxDynamicSharedMemorySize, smem_bytes);
  attn_fused<<<B_ * H_ * (S_ / 16), 256, smem_bytes, stream>>>(
      Qh, Kh, Vth, gmod, attn_out, ctxh);

  // output projection (f32 straight into d_out)
  gemm_wmma<3><<<ggrid, 256, 0, stream>>>(ctxh, WoT, bo, d_out, B_ * S_, D_, D_);
}
